// CausalTokenizer_61649960566816
// MI455X (gfx1250) — matrix operand order repaired and checked by reading, not compiled, not run on hardware
//
#include <hip/hip_runtime.h>
#include <hip/hip_bf16.h>
#include <math.h>

// ---------------- problem constants ----------------
#define B_    4
#define T_    8
#define C_    3
#define IMG_  224
#define P_    16
#define E_    512
#define NH_   8
#define NL_   128
#define LD_   128
#define G_    (IMG_/P_)        // 14
#define N_    (G_*G_)          // 196
#define PD_   (C_*P_*P_)       // 768
#define MLP_  (4*E_)           // 2048
#define S_    (NL_+N_)         // 324
#define DH_   (E_/NH_)         // 64
#define BT_   (B_*T_)          // 32
#define ROWS_ (BT_*S_)         // 10368

typedef __attribute__((ext_vector_type(16))) __bf16         v16bf;
typedef __attribute__((ext_vector_type(8)))  float          v8f;
typedef __attribute__((ext_vector_type(4)))  unsigned int   u32x4;
typedef __attribute__((ext_vector_type(8)))  unsigned short u16x8;
typedef __attribute__((ext_vector_type(8)))  int            i32x8;
typedef __attribute__((ext_vector_type(4)))  int            i32x4;

typedef unsigned short bf16_t;   // raw bf16 bits

#if defined(__has_builtin)
#if __has_builtin(__builtin_amdgcn_tensor_load_to_lds) && __has_builtin(__builtin_amdgcn_s_wait_tensorcnt)
#define HAVE_TDM 1
#endif
#endif

__device__ __forceinline__ bf16_t f2bf(float f) {
    unsigned x = __float_as_uint(f);
    unsigned r = x + 0x7FFFu + ((x >> 16) & 1u);   // round-to-nearest-even
    return (bf16_t)(r >> 16);
}
__device__ __forceinline__ float gelu_t(float x) {
    float inner = 0.7978845608028654f * (x + 0.044715f * x * x * x);
    return 0.5f * x * (1.0f + tanhf(inner));
}

#define WMMA_BF16(A_, B_v, C_) \
    __builtin_amdgcn_wmma_f32_16x16x32_bf16(false, (A_), false, (B_v), (short)0, (C_), false, false)

#ifdef HAVE_TDM
// LDS byte offset of a __shared__-derived pointer (low 32 bits of generic addr)
__device__ __forceinline__ unsigned lds_off(const void* p) {
    return (unsigned)(unsigned long long)(uintptr_t)p;
}
// TDM: DMA a 2D tile (64 rows x 32 cols of 2-byte elements, row stride K elems)
// from global into LDS. D# per CDNA5 ISA (groups 0/1; 2D so
// trailing groups zero). Tracked by TENSORcnt. 6-arg builtin form (clang-23 /
// amdgpu-toolchain): (uint32x4, int32x8, int32x4, int32x4, int32x8, i32 cpol).
__device__ __forceinline__ void tdm_load_b_tile(const bf16_t* gsrc, unsigned ldsa, int K)
{
    unsigned long long ga = (unsigned long long)(uintptr_t)gsrc;
    u32x4 g0;
    g0[0] = 1u;                                               // count=1 (valid user D#)
    g0[1] = ldsa;                                             // lds_addr
    g0[2] = (unsigned)(ga & 0xFFFFFFFFu);                     // global_addr[31:0]
    g0[3] = (unsigned)((ga >> 32) & 0x1FFFFFFu) | (2u << 30); // global_addr[56:32] | type=2
    i32x8 g1;
    g1[0] = 1 << 16;                                          // data_size=1 (2 bytes)
    g1[1] = (int)(((unsigned)K & 0xFFFFu) << 16);             // tensor_dim0[15:0] = K
    g1[2] = (int)((((unsigned)K >> 16) & 0xFFFFu) | (64u << 16)); // td0 hi | tensor_dim1=64
    g1[3] = (int)(32u << 16);                                 // td1 hi(0) | tile_dim0=32
    g1[4] = 64;                                               // tile_dim1=64 | tile_dim2=0
    g1[5] = K;                                                // tensor_dim0_stride[31:0]
    g1[6] = 0;                                                // stride hi | dim1_stride lo
    g1[7] = 0;
    i32x4 z4 = { 0, 0, 0, 0 };
    i32x8 z8 = { 0, 0, 0, 0, 0, 0, 0, 0 };
    __builtin_amdgcn_tensor_load_to_lds(g0, g1, z4, z4, z8, 0);
}
#endif

// ---------------- WMMA GEMM ---------------------------------------------------
// C = act(A_bf16[M][K] * Bt_bf16[N][K]^T + bias) + add ; dual outputs (f32/bf16)
// Block tile 128x64, 8 waves, each wave computes 32x32 (4 WMMA accs,
// 2 A-frags x 2 B-frags). Double-buffered LDS; A staged via registers,
// B staged via TDM tensor_load_to_lds (wave 0 issues, TENSORcnt + barrier
// publishes). One barrier per K-step.
// Requires M%128==0, N%64==0, K%32==0 (true for every GEMM in this model).
// act: 0=none, 1=gelu(tanh approx), 2=tanh
__global__ __launch_bounds__(256)
void gemm_wmma(const bf16_t* __restrict__ A,    // [M][K] bf16
               const bf16_t* __restrict__ Bt,   // [N][K] bf16 (pre-transposed weights)
               const float*  __restrict__ bias, // [N] or null
               const float*  __restrict__ addr_,// residual [M][N] f32 or null
               float*        __restrict__ Cf,   // f32 output or null
               bf16_t*       __restrict__ Cbf,  // bf16 output or null
               int M, int K, int Nn, int act)
{
    __shared__ __align__(16) bf16_t As[2][128][32];   // [buf][m][k]  16 KB
    __shared__ __align__(16) bf16_t Bs[2][64][32];    // [buf][n][k]   8 KB

    const int bm   = blockIdx.y * 128;
    const int bn   = blockIdx.x * 64;
    const int tid  = threadIdx.x;
    const int wave = tid >> 5;
    const int lane = tid & 31;
    const int hf   = lane >> 4;          // lane half (0/1)
    const int l15  = lane & 15;
    const int wr   = (wave >> 1) * 32;   // wave row group: 0,32,64,96
    const int wc   = (wave & 1) * 32;    // wave col group: 0,32

    // staging coordinates: 16B chunks; A needs 2 chunks/thread
    const int sr0 = tid >> 2;            // 0..63
    const int sr1 = sr0 + 64;            // 64..127
    const int sk  = (tid & 3) * 8;       // 0,8,16,24

    const bf16_t* Ap0 = &A[(size_t)(bm + sr0) * K + sk];
    const bf16_t* Ap1 = &A[(size_t)(bm + sr1) * K + sk];
#ifndef HAVE_TDM
    const bf16_t* Bp  = &Bt[(size_t)(bn + sr0) * K + sk];
#endif

    // preload tile 0
    u16x8 ra0 = *(const u16x8*)Ap0;
    u16x8 ra1 = *(const u16x8*)Ap1;
    *(u16x8*)&As[0][sr0][sk] = ra0;
    *(u16x8*)&As[0][sr1][sk] = ra1;
#ifdef HAVE_TDM
    if (wave == 0) {
        tdm_load_b_tile(&Bt[(size_t)bn * K], lds_off(&Bs[0][0][0]), K);
        __builtin_amdgcn_s_wait_tensorcnt((short)0);
    }
#else
    u16x8 rb = *(const u16x8*)Bp;
    *(u16x8*)&Bs[0][sr0][sk] = rb;
#endif
    __syncthreads();

    v8f acc00 = {}, acc01 = {}, acc10 = {}, acc11 = {};
    const int nk = K >> 5;

    for (int t = 0; t < nk; ++t) {
        const int cur = t & 1;
        const int nxt = cur ^ 1;
        // issue next tile's loads early (latency hides under WMMAs)
        if (t + 1 < nk) {
            ra0 = *(const u16x8*)(Ap0 + (size_t)(t + 1) * 32);
            ra1 = *(const u16x8*)(Ap1 + (size_t)(t + 1) * 32);
#ifdef HAVE_TDM
            if (wave == 0)
                tdm_load_b_tile(&Bt[(size_t)bn * K + (size_t)(t + 1) * 32],
                                lds_off(&Bs[nxt][0][0]), K);
#else
            rb  = *(const u16x8*)(Bp + (size_t)(t + 1) * 32);
#endif
        }
        if (t + 2 < nk)   // global_prefetch_b8 hint two tiles ahead
            __builtin_prefetch(Ap0 + (size_t)(t + 2) * 32, 0, 1);

        // A fragment: lane half 0 -> K 0..7 & 16..23 ; half 1 -> K 8..15 & 24..31
        union { v16bf v; u32x4 q[2]; } fa0, fa1, fb0, fb1;
        fa0.q[0] = *(const u32x4*)&As[cur][wr + l15][hf * 8];
        fa0.q[1] = *(const u32x4*)&As[cur][wr + l15][16 + hf * 8];
        fa1.q[0] = *(const u32x4*)&As[cur][wr + 16 + l15][hf * 8];
        fa1.q[1] = *(const u32x4*)&As[cur][wr + 16 + l15][16 + hf * 8];
        // B fragments: lane half 0 -> K 0..7 & 16..23 ; half 1 -> K 8..15 & 24..31
        fb0.q[0] = *(const u32x4*)&Bs[cur][wc + l15][hf * 8];
        fb0.q[1] = *(const u32x4*)&Bs[cur][wc + l15][16 + hf * 8];
        fb1.q[0] = *(const u32x4*)&Bs[cur][wc + 16 + l15][hf * 8];
        fb1.q[1] = *(const u32x4*)&Bs[cur][wc + 16 + l15][16 + hf * 8];

        acc00 = WMMA_BF16(fa0.v, fb0.v, acc00);
        acc01 = WMMA_BF16(fa0.v, fb1.v, acc01);
        acc10 = WMMA_BF16(fa1.v, fb0.v, acc10);
        acc11 = WMMA_BF16(fa1.v, fb1.v, acc11);

        // stash next tile into the alternate LDS buffer
        if (t + 1 < nk) {
            *(u16x8*)&As[nxt][sr0][sk] = ra0;
            *(u16x8*)&As[nxt][sr1][sk] = ra1;
#ifdef HAVE_TDM
            if (wave == 0) __builtin_amdgcn_s_wait_tensorcnt((short)0);
#else
            *(u16x8*)&Bs[nxt][sr0][sk] = rb;
#endif
        }
        __syncthreads();
    }

    // epilogue: C/D layout -> VGPR v: row = wr(+16) + hf*8 + v ; col = wc + l15 (+16)
    #pragma unroll
    for (int v = 0; v < 8; ++v) {
        int gm0 = bm + wr + hf * 8 + v;
        int gm1 = gm0 + 16;
        int gn0 = bn + wc + l15;
        int gn1 = gn0 + 16;
        float o00 = acc00[v], o01 = acc01[v], o10 = acc10[v], o11 = acc11[v];
        if (bias) {
            float b0 = bias[gn0], b1 = bias[gn1];
            o00 += b0; o01 += b1; o10 += b0; o11 += b1;
        }
        if (act == 1) {
            o00 = gelu_t(o00); o01 = gelu_t(o01); o10 = gelu_t(o10); o11 = gelu_t(o11);
        } else if (act == 2) {
            o00 = tanhf(o00); o01 = tanhf(o01); o10 = tanhf(o10); o11 = tanhf(o11);
        }
        if (addr_) {
            o00 += addr_[(size_t)gm0 * Nn + gn0];
            o01 += addr_[(size_t)gm0 * Nn + gn1];
            o10 += addr_[(size_t)gm1 * Nn + gn0];
            o11 += addr_[(size_t)gm1 * Nn + gn1];
        }
        if (Cf) {
            Cf[(size_t)gm0 * Nn + gn0] = o00;
            Cf[(size_t)gm0 * Nn + gn1] = o01;
            Cf[(size_t)gm1 * Nn + gn0] = o10;
            Cf[(size_t)gm1 * Nn + gn1] = o11;
        }
        if (Cbf) {
            Cbf[(size_t)gm0 * Nn + gn0] = f2bf(o00);
            Cbf[(size_t)gm0 * Nn + gn1] = f2bf(o01);
            Cbf[(size_t)gm1 * Nn + gn0] = f2bf(o10);
            Cbf[(size_t)gm1 * Nn + gn1] = f2bf(o11);
        }
    }
}

// ---------------- weight prep: [K][N] f32 -> [N][K] bf16 ----------------------
__global__ void wtrans_k(const float* __restrict__ src, bf16_t* __restrict__ dst,
                         int K, int N, int total)
{
    for (int o = blockIdx.x * blockDim.x + threadIdx.x; o < total; o += gridDim.x * blockDim.x) {
        int n = o % N, k = o / N;
        dst[(size_t)n * K + k] = f2bf(src[o]);
    }
}

// ---------------- LayerNorm over E=512, one block per row, bf16 out ----------
__global__ __launch_bounds__(256)
void layernorm_k(const float* __restrict__ X, const float* __restrict__ s,
                 const float* __restrict__ b, bf16_t* __restrict__ Y)
{
    __shared__ float red[256];
    const int row = blockIdx.x;
    const int t = threadIdx.x;
    const float* x = X + (size_t)row * E_;
    float v0 = x[t], v1 = x[t + 256];

    red[t] = v0 + v1;
    __syncthreads();
    for (int o = 128; o > 0; o >>= 1) { if (t < o) red[t] += red[t + o]; __syncthreads(); }
    float mean = red[0] * (1.0f / E_);
    __syncthreads();

    float d0 = v0 - mean, d1 = v1 - mean;
    red[t] = d0 * d0 + d1 * d1;
    __syncthreads();
    for (int o = 128; o > 0; o >>= 1) { if (t < o) red[t] += red[t + o]; __syncthreads(); }
    float rstd = rsqrtf(red[0] * (1.0f / E_) + 1e-5f);

    bf16_t* y = Y + (size_t)row * E_;
    y[t]       = f2bf(d0 * rstd * s[t]       + b[t]);
    y[t + 256] = f2bf(d1 * rstd * s[t + 256] + b[t + 256]);
}

// ---------------- space attention: one block per (bt, head, q-row) ----------
// maskmode 0 = encoder (patches can't see latents), 1 = decoder (latents can't see patches)
__global__ __launch_bounds__(128)
void attn_space(const float* __restrict__ qkv, bf16_t* __restrict__ out, int maskmode)
{
    int id = blockIdx.x;
    const int q  = id % S_; id /= S_;
    const int h  = id % NH_; id /= NH_;
    const int bt = id;

    __shared__ float sc[S_];
    __shared__ float red[128];
    const size_t base = (size_t)bt * S_ * 3 * E_;
    const float* qp = qkv + base + (size_t)q * 3 * E_ + h * DH_;

    float lmax = -1e30f;
    for (int k = threadIdx.x; k < S_; k += 128) {
        const float* kp = qkv + base + (size_t)k * 3 * E_ + E_ + h * DH_;
        float sv = 0.f;
        for (int d = 0; d < DH_; ++d) sv += qp[d] * kp[d];
        sv *= 0.125f; // 1/sqrt(64)
        bool allowed = (maskmode == 0) ? !(q >= NL_ && k < NL_)
                                       : !(q <  NL_ && k >= NL_);
        if (!allowed) sv = -1e9f;
        sc[k] = sv;
        lmax = fmaxf(lmax, sv);
    }
    red[threadIdx.x] = lmax; __syncthreads();
    for (int o = 64; o > 0; o >>= 1) { if (threadIdx.x < o) red[threadIdx.x] = fmaxf(red[threadIdx.x], red[threadIdx.x + o]); __syncthreads(); }
    float mx = red[0]; __syncthreads();

    float lsum = 0.f;
    for (int k = threadIdx.x; k < S_; k += 128) { float e = expf(sc[k] - mx); sc[k] = e; lsum += e; }
    red[threadIdx.x] = lsum; __syncthreads();
    for (int o = 64; o > 0; o >>= 1) { if (threadIdx.x < o) red[threadIdx.x] += red[threadIdx.x + o]; __syncthreads(); }
    float inv = 1.f / red[0];
    __syncthreads();

    if (threadIdx.x < DH_) {
        int d = threadIdx.x;
        float o = 0.f;
        for (int k = 0; k < S_; ++k)
            o += sc[k] * qkv[base + (size_t)k * 3 * E_ + 2 * E_ + h * DH_ + d];
        out[(size_t)(bt * S_ + q) * E_ + h * DH_ + d] = f2bf(o * inv);
    }
}

// ---------------- causal time attention over T=8: one block per (b, s, h) ---
__global__ __launch_bounds__(64)
void attn_time(const float* __restrict__ qkv, bf16_t* __restrict__ out)
{
    int id = blockIdx.x;
    const int h = id % NH_; id /= NH_;
    const int s = id % S_;  id /= S_;
    const int b = id;
    const int qt = threadIdx.x;
    if (qt >= T_) return;

    float sc[T_];
    const float* qp = qkv + (size_t)((b * T_ + qt) * S_ + s) * 3 * E_ + h * DH_;
    float mx = -1e30f;
    for (int kt = 0; kt <= qt; ++kt) {
        const float* kp = qkv + (size_t)((b * T_ + kt) * S_ + s) * 3 * E_ + E_ + h * DH_;
        float v = 0.f;
        for (int d = 0; d < DH_; ++d) v += qp[d] * kp[d];
        v *= 0.125f;
        sc[kt] = v; mx = fmaxf(mx, v);
    }
    float sum = 0.f;
    for (int kt = 0; kt <= qt; ++kt) { sc[kt] = expf(sc[kt] - mx); sum += sc[kt]; }
    float inv = 1.f / sum;
    for (int d = 0; d < DH_; ++d) {
        float o = 0.f;
        for (int kt = 0; kt <= qt; ++kt)
            o += sc[kt] * qkv[(size_t)((b * T_ + kt) * S_ + s) * 3 * E_ + 2 * E_ + h * DH_ + d];
        out[(size_t)((b * T_ + qt) * S_ + s) * E_ + h * DH_ + d] = f2bf(o * inv);
    }
}

// ---------------- random-mask (rank == stable argsort position) -------------
__global__ __launch_bounds__(256)
void mask_k(const float* __restrict__ noise, const float* __restrict__ ratios,
            float* __restrict__ maskout)
{
    const int bt = blockIdx.x;
    const int b  = bt / T_;
    __shared__ float nr[N_];
    for (int i = threadIdx.x; i < N_; i += 256) nr[i] = noise[(size_t)bt * N_ + i];
    __syncthreads();
    int len_keep = (int)(N_ * (1.0f - ratios[b] * 0.9f));
    for (int j = threadIdx.x; j < N_; j += 256) {
        float nj = nr[j];
        int rank = 0;
        for (int i = 0; i < N_; ++i) {
            float ni = nr[i];
            if (ni < nj || (ni == nj && i < j)) rank++;
        }
        maskout[(size_t)bt * N_ + j] = (rank < len_keep) ? 1.0f : 0.0f;
    }
}

// ---------------- patchify (bf16 out) / unpatchify ---------------------------
__global__ void patchify_k(const float* __restrict__ x, bf16_t* __restrict__ xp, int total)
{
    for (int o = blockIdx.x * blockDim.x + threadIdx.x; o < total; o += gridDim.x * blockDim.x) {
        int pd = o % PD_;   int r = o / PD_;
        int n  = r % N_;    int bt = r / N_;
        int p1 = pd / (P_ * C_); int rem = pd % (P_ * C_);
        int p2 = rem / C_;  int c = rem % C_;
        int gh = n / G_, gw = n % G_;
        size_t xi = (((size_t)bt * C_ + c) * IMG_ + (gh * P_ + p1)) * IMG_ + (gw * P_ + p2);
        xp[o] = f2bf(x[xi]);
    }
}

__global__ void unpatchify_k(const float* __restrict__ rp, float* __restrict__ recon, int total)
{
    for (int o = blockIdx.x * blockDim.x + threadIdx.x; o < total; o += gridDim.x * blockDim.x) {
        int j  = o % IMG_;
        int i  = (o / IMG_) % IMG_;
        int c  = (o / (IMG_ * IMG_)) % C_;
        int bt = o / (IMG_ * IMG_ * C_);
        int gh = i / P_, p1 = i % P_, gw = j / P_, p2 = j % P_;
        recon[o] = rp[((size_t)bt * N_ + (gh * G_ + gw)) * PD_ + (p1 * P_ + p2) * C_ + c];
    }
}

// ---------------- sequence assembly -----------------------------------------
__global__ void build_enc_in_k(const float* __restrict__ patches, const float* __restrict__ maskv,
                               const float* __restrict__ latent_tokens,
                               const float* __restrict__ mask_token,
                               float* __restrict__ encin, int total)
{
    for (int o = blockIdx.x * blockDim.x + threadIdx.x; o < total; o += gridDim.x * blockDim.x) {
        int e = o % E_; int r = o / E_;
        int s = r % S_; int bt = r / S_;
        float v;
        if (s < NL_) v = latent_tokens[s * E_ + e];
        else {
            int n = s - NL_;
            float m = maskv[(size_t)bt * N_ + n];
            v = patches[((size_t)bt * N_ + n) * E_ + e] * m + mask_token[e] * (1.f - m);
        }
        encin[o] = v;
    }
}

__global__ void build_dec_in_k(const float* __restrict__ lat, const float* __restrict__ dec_tokens,
                               float* __restrict__ decin, int total)
{
    for (int o = blockIdx.x * blockDim.x + threadIdx.x; o < total; o += gridDim.x * blockDim.x) {
        int e = o % E_; int r = o / E_;
        int s = r % S_; int bt = r / S_;
        decin[o] = (s < NL_) ? lat[((size_t)bt * NL_ + s) * E_ + e]
                             : dec_tokens[(s - NL_) * E_ + e];
    }
}

__global__ void gather_rows_k(const float* __restrict__ X, bf16_t* __restrict__ Y,
                              int s0, int cnt, int total)
{
    for (int o = blockIdx.x * blockDim.x + threadIdx.x; o < total; o += gridDim.x * blockDim.x) {
        int e = o % E_; int r = o / E_;
        int s = r % cnt; int bt = r / cnt;
        Y[o] = f2bf(X[((size_t)bt * S_ + s0 + s) * E_ + e]);
    }
}

// ---------------- host-side orchestration -----------------------------------
struct Blk {
    const float *ln1_s, *ln1_b, *ln2_s, *ln2_b;
    const float *qkv_w, *qkv_b, *proj_w, *proj_b, *fc1_w, *fc1_b, *fc2_w, *fc2_b;
};
struct BlkW { const bf16_t *qkv, *proj, *fc1, *fc2; };

static inline void launch_gemm(const bf16_t* A, const bf16_t* Bt, const float* bias,
                               const float* add, float* Cf, bf16_t* Cbf,
                               int M, int K, int Nn, int act, hipStream_t stream)
{
    dim3 grid(Nn / 64, M / 128);
    gemm_wmma<<<grid, dim3(256), 0, stream>>>(A, Bt, bias, add, Cf, Cbf, M, K, Nn, act);
}

extern "C" void kernel_launch(void* const* d_in, const int* in_sizes, int n_in,
                              void* d_out, int out_size, void* d_ws, size_t ws_size,
                              hipStream_t stream)
{
    (void)in_sizes; (void)n_in; (void)out_size; (void)ws_size;
    auto F = [&](int i) { return (const float*)d_in[i]; };

    // input order: x, noise, mask_ratios, then params leaves in insertion order
    const float* x_in   = F(0);
    const float* noise  = F(1);
    const float* ratios = F(2);
    const float* pe_w   = F(3);
    const float* pe_b   = F(4);
    const float* mask_token    = F(5);
    const float* latent_tokens = F(6);
    const float* dec_tokens    = F(7);
    Blk enc[4], dec[4];
    for (int i = 0; i < 4; ++i) {
        int base = 8 + i * 12;
        enc[i] = { F(base+0), F(base+1), F(base+2), F(base+3), F(base+4), F(base+5),
                   F(base+6), F(base+7), F(base+8), F(base+9), F(base+10), F(base+11) };
    }
    const float* to_lat_w   = F(56);
    const float* to_lat_b   = F(57);
    const float* from_lat_w = F(58);
    const float* from_lat_b = F(59);
    for (int i = 0; i < 4; ++i) {
        int base = 60 + i * 12;
        dec[i] = { F(base+0), F(base+1), F(base+2), F(base+3), F(base+4), F(base+5),
                   F(base+6), F(base+7), F(base+8), F(base+9), F(base+10), F(base+11) };
    }
    const float* to_pix_w = F(108);
    const float* to_pix_b = F(109);

    // output layout (flat, return order): z, recon, mask, patches
    float* out         = (float*)d_out;
    float* z_out       = out;                                   // 4*8*128*128
    float* recon_out   = z_out + (size_t)BT_ * NL_ * LD_;       // 4*8*3*224*224
    float* mask_out    = recon_out + (size_t)BT_ * C_ * IMG_ * IMG_;
    float* patches_out = mask_out + (size_t)BT_ * N_;           // 4*8*196*512

    // ---- workspace layout: f32 region first, then bf16 region ----
    float* wf = (float*)d_ws;
    float* xbuf   = wf;  wf += (size_t)ROWS_ * E_;
    float* x2buf  = wf;  wf += (size_t)ROWS_ * E_;
    float* qkvbuf = wf;  wf += (size_t)ROWS_ * 3 * E_;
    float* latbuf = wf;  wf += (size_t)BT_ * NL_ * E_;
    float* rp     = wf;  wf += (size_t)BT_ * N_ * PD_;

    bf16_t* wb = (bf16_t*)wf;
    bf16_t* xp_bf   = wb;  wb += (size_t)BT_ * N_ * PD_;
    bf16_t* hbuf_bf = wb;  wb += (size_t)ROWS_ * E_;        // LN out / pixel gather
    bf16_t* abuf_bf = wb;  wb += (size_t)ROWS_ * E_;
    bf16_t* mlp_bf  = wb;  wb += (size_t)ROWS_ * MLP_;
    bf16_t* latc_bf = wb;  wb += (size_t)BT_ * NL_ * E_;
    bf16_t* z_bf    = wb;  wb += (size_t)BT_ * NL_ * LD_;
    bf16_t* wpool   = wb;  // transposed bf16 weights appended here

    auto blocks_for = [](int total) { return (total + 255) / 256; };

    // ---- one-time (per launch) weight convert+transpose -> [N][K] bf16 ----
    bf16_t* wcur = wpool;
    auto prep_w = [&](const float* src, int K, int N) -> const bf16_t* {
        bf16_t* dst = wcur; wcur += (size_t)K * N;
        wtrans_k<<<blocks_for(K * N), 256, 0, stream>>>(src, dst, K, N, K * N);
        return dst;
    };
    const bf16_t* pe_wt       = prep_w(pe_w,       PD_, E_);
    const bf16_t* to_lat_wt   = prep_w(to_lat_w,   E_,  LD_);
    const bf16_t* from_lat_wt = prep_w(from_lat_w, LD_, E_);
    const bf16_t* to_pix_wt   = prep_w(to_pix_w,   E_,  PD_);
    BlkW encw[4], decw[4];
    for (int i = 0; i < 4; ++i)
        encw[i] = { prep_w(enc[i].qkv_w, E_, 3 * E_), prep_w(enc[i].proj_w, E_, E_),
                    prep_w(enc[i].fc1_w, E_, MLP_),   prep_w(enc[i].fc2_w, MLP_, E_) };
    for (int i = 0; i < 4; ++i)
        decw[i] = { prep_w(dec[i].qkv_w, E_, 3 * E_), prep_w(dec[i].proj_w, E_, E_),
                    prep_w(dec[i].fc1_w, E_, MLP_),   prep_w(dec[i].fc2_w, MLP_, E_) };

    // ---- patchify + mask + patch embed ----
    int tot_patch = BT_ * N_ * PD_;
    patchify_k<<<blocks_for(tot_patch), 256, 0, stream>>>(x_in, xp_bf, tot_patch);
    mask_k<<<BT_, 256, 0, stream>>>(noise, ratios, mask_out);
    launch_gemm(xp_bf, pe_wt, pe_b, nullptr, patches_out, nullptr,
                BT_ * N_, PD_, E_, 0, stream);

    // ---- encoder input ----
    int tot_seq = ROWS_ * E_;
    build_enc_in_k<<<blocks_for(tot_seq), 256, 0, stream>>>(patches_out, mask_out,
                                                            latent_tokens, mask_token,
                                                            xbuf, tot_seq);

    auto run_block = [&](const Blk& p, const BlkW& pw, int mode) {
        layernorm_k<<<ROWS_, 256, 0, stream>>>(xbuf, p.ln1_s, p.ln1_b, hbuf_bf);
        launch_gemm(hbuf_bf, pw.qkv, p.qkv_b, nullptr, qkvbuf, nullptr,
                    ROWS_, E_, 3 * E_, 0, stream);
        if (mode == 2)
            attn_time<<<B_ * S_ * NH_, 64, 0, stream>>>(qkvbuf, abuf_bf);
        else
            attn_space<<<BT_ * NH_ * S_, 128, 0, stream>>>(qkvbuf, abuf_bf, mode);
        launch_gemm(abuf_bf, pw.proj, p.proj_b, xbuf, x2buf, nullptr,
                    ROWS_, E_, E_, 0, stream);
        layernorm_k<<<ROWS_, 256, 0, stream>>>(x2buf, p.ln2_s, p.ln2_b, hbuf_bf);
        launch_gemm(hbuf_bf, pw.fc1, p.fc1_b, nullptr, nullptr, mlp_bf,
                    ROWS_, E_, MLP_, 1, stream);
        launch_gemm(mlp_bf, pw.fc2, p.fc2_b, x2buf, xbuf, nullptr,
                    ROWS_, MLP_, E_, 0, stream);
    };

    // ---- encoder: 3 space blocks (enc mask) + 1 time block ----
    for (int i = 0; i < 3; ++i) run_block(enc[i], encw[i], 0);
    run_block(enc[3], encw[3], 2);

    // ---- to latent (tanh) -> z (f32 out + bf16 copy) ; from latent ----
    int tot_lat = BT_ * NL_ * E_;
    gather_rows_k<<<blocks_for(tot_lat), 256, 0, stream>>>(xbuf, latc_bf, 0, NL_, tot_lat);
    launch_gemm(latc_bf, to_lat_wt, to_lat_b, nullptr, z_out, z_bf,
                BT_ * NL_, E_, LD_, 2, stream);
    launch_gemm(z_bf, from_lat_wt, from_lat_b, nullptr, latbuf, nullptr,
                BT_ * NL_, LD_, E_, 0, stream);

    // ---- decoder input ----
    build_dec_in_k<<<blocks_for(tot_seq), 256, 0, stream>>>(latbuf, dec_tokens, xbuf, tot_seq);

    // ---- decoder: 3 space blocks (dec mask) + 1 time block ----
    for (int i = 0; i < 3; ++i) run_block(dec[i], decw[i], 1);
    run_block(dec[3], decw[3], 2);

    // ---- to pixels + unpatchify ----
    int tot_pix_rows = BT_ * N_ * E_;
    gather_rows_k<<<blocks_for(tot_pix_rows), 256, 0, stream>>>(xbuf, hbuf_bf, NL_, N_, tot_pix_rows);
    launch_gemm(hbuf_bf, to_pix_wt, to_pix_b, nullptr, rp, nullptr,
                BT_ * N_, E_, PD_, 0, stream);
    int tot_img = BT_ * C_ * IMG_ * IMG_;
    unpatchify_k<<<blocks_for(tot_img), 256, 0, stream>>>(rp, recon_out, tot_img);
}
